// RelativePositionEmbedding_68719477350
// MI455X (gfx1250) — matrix operand order repaired and checked by reading, not compiled, not run on hardware
//
#include <hip/hip_runtime.h>
#include <hip/hip_bf16.h>
#include <math.h>

// ---------------------------------------------------------------------------
// RelativePositionEmbedding for MI455X (gfx1250), wave32 + WMMA bf16 +
// async direct-to-LDS staging. B=2, N=256, D=256 fixed by the reference.
// ---------------------------------------------------------------------------

typedef __attribute__((ext_vector_type(16))) __bf16 v16bf;
typedef __attribute__((ext_vector_type(8)))  float  v8f;

union FragU { v16bf v; uint4 q[2]; };

__device__ __forceinline__ unsigned f2bf(float f) {
    unsigned u = __float_as_uint(f);
    u += 0x7FFFu + ((u >> 16) & 1u);   // round-to-nearest-even
    return u >> 16;
}

__device__ __forceinline__ v16bf ld16(const unsigned short* p0,
                                      const unsigned short* p1) {
    FragU u;
    u.q[0] = *reinterpret_cast<const uint4*>(p0);
    u.q[1] = *reinterpret_cast<const uint4*>(p1);
    return u.v;
}

// 16-byte global -> LDS async copy (ASYNCcnt-tracked, no VGPR round trip)
__device__ __forceinline__ void async_cp16(void* dst_lds, const void* src_glb) {
    auto* d3 = (__attribute__((address_space(3))) unsigned char*)dst_lds;
    auto* s1 = (const __attribute__((address_space(1))) unsigned char*)src_glb;
    asm volatile("global_load_async_to_lds_b128 %0, %1, off"
                 :: "v"(d3), "v"(s1) : "memory");
}

__device__ __forceinline__ void wait_async0() {
    asm volatile("s_wait_asynccnt 0x0" ::: "memory");
}

// ------------------------- prep: f32 -> bf16 weights ------------------------
__global__ void prep_kernel(const float* __restrict__ Wd,
                            const float* __restrict__ Wa,
                            const float* __restrict__ bd,
                            const float* __restrict__ ba,
                            unsigned short* __restrict__ gWd,
                            unsigned short* __restrict__ gWa,
                            float* __restrict__ gbs) {
    const int i = (blockIdx.x * 256 + threadIdx.x) * 2;   // 32768 threads * 2
    const float2 wd = *reinterpret_cast<const float2*>(Wd + i);
    const float2 wa = *reinterpret_cast<const float2*>(Wa + i);
    *reinterpret_cast<unsigned*>(gWd + i) = f2bf(wd.x) | (f2bf(wd.y) << 16);
    *reinterpret_cast<unsigned*>(gWa + i) = f2bf(wa.x) | (f2bf(wa.y) << 16);
    if (blockIdx.x == 0) gbs[threadIdx.x] = bd[threadIdx.x] + ba[threadIdx.x];
}

// ------------------------------ kNN kernel ---------------------------------
__global__ void knn_kernel(const float* __restrict__ points,
                           int* __restrict__ knn) {
    __shared__ float px[256], py[256], pz[256];
    const int b = blockIdx.x;
    const int n = threadIdx.x;
    const float* pb = points + b * 768;
    px[n] = pb[n * 3 + 0];
    py[n] = pb[n * 3 + 1];
    pz[n] = pb[n * 3 + 2];
    __syncthreads();
    const float qx = px[n], qy = py[n], qz = pz[n];
    float d0 = 3.0e38f, d1 = 3.0e38f, d2 = 3.0e38f, d3 = 3.0e38f;
    int   i0 = 0, i1 = 0, i2 = 0, i3 = 0;
    for (int m = 0; m < 256; ++m) {
        const float dx = px[m] - qx, dy = py[m] - qy, dz = pz[m] - qz;
        const float dd = dx * dx + dy * dy + dz * dz;
        if (dd < d0)      { d3=d2;i3=i2; d2=d1;i2=i1; d1=d0;i1=i0; d0=dd;i0=m; }
        else if (dd < d1) { d3=d2;i3=i2; d2=d1;i2=i1; d1=dd;i1=m; }
        else if (dd < d2) { d3=d2;i3=i2; d2=dd;i2=m; }
        else if (dd < d3) { d3=dd;i3=m; }
    }
    // reference drops the first hit (self, dist 0) and keeps the next 3
    knn[(b * 256 + n) * 3 + 0] = i1;
    knn[(b * 256 + n) * 3 + 1] = i2;
    knn[(b * 256 + n) * 3 + 2] = i3;
}

// ------------------------------ main kernel --------------------------------
// LDS layout (dynamic):
#define OFF_WD   0          // 65536 bf16  (Wd row-major)      131072 B
#define OFF_WA   131072     // 65536 bf16  (Wa row-major)      131072 B
#define OFF_AB   262144     // 64x256 bf16 embedding tile       32768 B
#define OFF_PTS  294912     // 256x3 float                       3072 B
#define OFF_DIV  297984     // 128 float div_term                 512 B
#define OFF_BS   298496     // 256 float (bd+ba)                 1024 B
#define SMEM_BYTES 299520

__launch_bounds__(512)
__global__ void rpe_main(const float* __restrict__ points,
                         const unsigned short* __restrict__ gWd,
                         const unsigned short* __restrict__ gWa,
                         const float* __restrict__ gbs,
                         const int*   __restrict__ knn,
                         float*       __restrict__ out) {
    extern __shared__ unsigned char smem[];
    unsigned short* sWd  = (unsigned short*)(smem + OFF_WD);
    unsigned short* sWa  = (unsigned short*)(smem + OFF_WA);
    unsigned short* sAb  = (unsigned short*)(smem + OFF_AB);
    float*          sPts = (float*)(smem + OFF_PTS);
    float*          sDiv = (float*)(smem + OFF_DIV);
    float*          sBs  = (float*)(smem + OFF_BS);

    const int tid = threadIdx.x;
    const int bn  = blockIdx.x;          // (b,n) pair, 0..511
    const int b   = bn >> 8;
    const int n   = bn & 255;

    // ---- async-stage 256KB of bf16 weights straight into LDS -----------
    #pragma unroll 4
    for (int i = tid * 8; i < 65536; i += 4096) {
        async_cp16(sWd + i, gWd + i);
        async_cp16(sWa + i, gWa + i);
    }
    // small tables via the normal path (overlaps with the async DMA)
    const float* pbase = points + b * 768;
    for (int i = tid; i < 768; i += 512) sPts[i] = pbase[i];
    for (int i = tid; i < 128; i += 512)
        sDiv[i] = __expf(-0.07195562243f * (float)i);   // exp(-2i*ln(1e4)/256)
    for (int i = tid; i < 256; i += 512) sBs[i] = gbs[i];
    const int k0 = knn[bn * 3 + 0];
    const int k1 = knn[bn * 3 + 1];
    const int k2 = knn[bn * 3 + 2];
    wait_async0();
    __syncthreads();

    const float pnx = sPts[n * 3 + 0];
    const float pny = sPts[n * 3 + 1];
    const float pnz = sPts[n * 3 + 2];
    // reference vectors to the 3 nearest neighbors
    const float r0x = sPts[k0*3+0]-pnx, r0y = sPts[k0*3+1]-pny, r0z = sPts[k0*3+2]-pnz;
    const float r1x = sPts[k1*3+0]-pnx, r1y = sPts[k1*3+1]-pny, r1z = sPts[k1*3+2]-pnz;
    const float r2x = sPts[k2*3+0]-pnx, r2y = sPts[k2*3+1]-pny, r2z = sPts[k2*3+2]-pnz;

    const int row  = tid >> 3;      // 0..63: 0-15 dist rows, 16-63 angle rows
    const int ch   = tid & 7;       // 32-wide k-chunk within the row
    const int mrow = row & 15;
    const int lane = tid & 31;
    const int wv   = tid >> 5;      // wave id == output column tile jt
    const int l15  = lane & 15;
    const int h    = lane >> 4;

    for (int mt = 0; mt < 16; ++mt) {
        // ---- build 64x256 bf16 embedding tile in LDS -------------------
        const int   mm = mt * 16 + mrow;
        const float ax = sPts[mm*3+0] - pnx;
        const float ay = sPts[mm*3+1] - pny;
        const float az = sPts[mm*3+2] - pnz;
        float x;
        if (row < 16) {
            x = sqrtf(ax*ax + ay*ay + az*az) * 5.0f;          // dist / SIGMA_D
        } else {
            const int q = (row >> 4) - 1;                     // neighbor id
            const float rx = (q == 0) ? r0x : ((q == 1) ? r1x : r2x);
            const float ry = (q == 0) ? r0y : ((q == 1) ? r1y : r2y);
            const float rz = (q == 0) ? r0z : ((q == 1) ? r1z : r2z);
            const float cx = ry*az - rz*ay;
            const float cy = rz*ax - rx*az;
            const float cz = rx*ay - ry*ax;
            const float sv = sqrtf(cx*cx + cy*cy + cz*cz);
            const float cv = rx*ax + ry*ay + rz*az;
            x = atan2f(sv, cv) * 3.81971863421f;              // FACTOR_A
        }
        unsigned* arow = reinterpret_cast<unsigned*>(sAb + row * 256 + ch * 32);
        #pragma unroll
        for (int f = 0; f < 16; ++f) {
            const float om = x * sDiv[ch * 16 + f];
            float s, c;
            __sincosf(om, &s, &c);
            arow[f] = f2bf(s) | (f2bf(c) << 16);              // [sin, cos] pair
        }
        __syncthreads();

        // ---- WMMA GEMMs: this wave owns output columns jt = wv ---------
        v8f accd = {};
        {
            const unsigned short* abase = sAb + l15 * 256;
            const unsigned short* bbase = sWd + (wv * 16 + l15) * 256;
            #pragma unroll
            for (int kt = 0; kt < 8; ++kt) {
                const v16bf A  = ld16(abase + 32*kt + 8*h,
                                      abase + 32*kt + 16 + 8*h);
                const v16bf Bf = ld16(bbase + 32*kt + 8*h,
                                      bbase + 32*kt + 16 + 8*h);
                accd = __builtin_amdgcn_wmma_f32_16x16x32_bf16(
                           false, A, false, Bf, (short)0, accd, false, false);
            }
        }
        v8f amax = {};
        {
            const unsigned short* bbase = sWa + (wv * 16 + l15) * 256;
            #pragma unroll
            for (int q = 0; q < 3; ++q) {
                v8f acc = {};
                const unsigned short* abase = sAb + (16 + q * 16 + l15) * 256;
                #pragma unroll
                for (int kt = 0; kt < 8; ++kt) {
                    const v16bf A  = ld16(abase + 32*kt + 8*h,
                                          abase + 32*kt + 16 + 8*h);
                    const v16bf Bf = ld16(bbase + 32*kt + 8*h,
                                          bbase + 32*kt + 16 + 8*h);
                    acc = __builtin_amdgcn_wmma_f32_16x16x32_bf16(
                              false, A, false, Bf, (short)0, acc, false, false);
                }
                if (q == 0) {
                    amax = acc;
                } else {
                    #pragma unroll
                    for (int r = 0; r < 8; ++r) amax[r] = fmaxf(amax[r], acc[r]);
                }
            }
        }
        // ---- epilogue: out[b,n,m,j] = d + bd + ba + max_k(a) ----------
        const float bsv = sBs[wv * 16 + l15];
        float* op = out + (size_t)bn * 65536
                        + (size_t)(mt * 16 + 8 * h) * 256
                        + (size_t)(wv * 16 + l15);
        #pragma unroll
        for (int r = 0; r < 8; ++r)
            op[(size_t)r * 256] = accd[r] + amax[r] + bsv;
        __syncthreads();   // before overwriting the embedding tile
    }
}

// ------------------------------ launcher -----------------------------------
// d_ws layout: [0,128K) Wd bf16 | [128K,256K) Wa bf16 | +1K bsum | +6K knn
extern "C" void kernel_launch(void* const* d_in, const int* in_sizes, int n_in,
                              void* d_out, int out_size, void* d_ws, size_t ws_size,
                              hipStream_t stream) {
    (void)in_sizes; (void)n_in; (void)out_size; (void)ws_size;
    const float* points = (const float*)d_in[0];
    const float* Wd     = (const float*)d_in[1];
    const float* bd     = (const float*)d_in[2];
    const float* Wa     = (const float*)d_in[3];
    const float* ba     = (const float*)d_in[4];
    float*       out    = (float*)d_out;

    unsigned char* ws = (unsigned char*)d_ws;
    unsigned short* gWd = (unsigned short*)(ws);
    unsigned short* gWa = (unsigned short*)(ws + 131072);
    float*          gbs = (float*)(ws + 262144);
    int*            knn = (int*)(ws + 263168);

    prep_kernel<<<dim3(128), dim3(256), 0, stream>>>(Wd, Wa, bd, ba, gWd, gWa, gbs);
    knn_kernel<<<dim3(2), dim3(256), 0, stream>>>(points, knn);
    rpe_main<<<dim3(512), dim3(512), SMEM_BYTES, stream>>>(
        points, gWd, gWa, gbs, knn, out);
}
